// LFFlow_39135742001768
// MI455X (gfx1250) — matrix operand order repaired and checked by reading, not compiled, not run on hardware
//
#include <hip/hip_runtime.h>

// ---------------------------------------------------------------------------
// MI455X (gfx1250) implementation: bf16 WMMA (16x16x32, f32 accum) for all
// edge/node MLPs, LDS restaging between layers, f32 atomics for segment_sum
// and forces. Wave32; one 16-row tile per wave; 4 waves / block.
// ---------------------------------------------------------------------------

typedef __bf16 bf16;
typedef __attribute__((ext_vector_type(16))) __bf16 v16bf;
typedef __attribute__((ext_vector_type(8)))  __bf16 v8bf;
typedef __attribute__((ext_vector_type(8)))  float  v8f;

constexpr int   N_NODES = 50000;
constexpr int   N_EDGES = 800000;
constexpr int   H       = 64;
constexpr int   T_ITER  = 3;
constexpr float BOX     = 10.0f;
constexpr float DT      = 0.01f;

constexpr int NH         = N_NODES * H;     // 3,200,000
constexpr int N3         = N_NODES * 3;
constexpr int NODE_TILES = N_NODES / 16;    // 3125
constexpr int EDGE_TILES = N_EDGES / 16;    // 50000
constexpr int WAVES      = 4;               // waves per block (wave32)

#define DEVFN __device__ __forceinline__

DEVFN float silu_f(float x) { return x / (1.0f + __expf(-x)); }

DEVFN v8f wmma_bf16(v16bf a, v16bf b, v8f c) {
  // D = A(16x32 bf16) x B(32x16 bf16) + C(16x16 f32)
  return __builtin_amdgcn_wmma_f32_16x16x32_bf16(
      /*neg_a=*/false, a, /*neg_b=*/false, b,
      /*c_mod=*/(short)0, c, /*reuse_a=*/false, /*reuse_b=*/false);
}

// A-operand fragment (ISA 7.12.2, 16-bit A 16x32): lane holds row m=lane&15;
// half hi=lane>>4 holds K = kb+hi*8..+7 (elems 0..7) and kb+16+hi*8..+7 (8..15).
// `row` points at a 64-wide bf16 row (global h_bf16 or an LDS activation tile).
DEVFN v16bf load_a_frag(const bf16* row, int kb64, int hi) {
  union { v16bf v; v8bf h[2]; } u;
  int k0 = kb64 + hi * 8;
  u.h[0] = *(const v8bf*)(row + k0);
  u.h[1] = *(const v8bf*)(row + k0 + 16);
  return u.v;
}

// B-operand fragment from pre-permuted weight buffer: frag (kc,nt) is stored
// as 32 lanes x 16 bf16 contiguous -> one 32B load per lane.
DEVFN v16bf load_b_frag(const bf16* wf, int kc, int nt, int lane) {
  return *(const v16bf*)(wf + ((size_t)((kc * 4 + nt) * 32 + lane)) * 16);
}

DEVFN void wait_lds() { asm volatile("s_wait_dscnt 0" ::: "memory"); }

// ---------------------------------------------------------------------------
// Weight conversion: f32 row-major [T,K,64] -> bf16 WMMA-B fragment order.
// dst[t*K*64 + ((kc*4+nt)*32+lane)*16 + j] = W[t][kc*32+(j>>3)*16+(lane>>4)*8+(j&7)][nt*16+(lane&15)]
// ---------------------------------------------------------------------------
__global__ void k_wconv_frag(const float* __restrict__ src, bf16* __restrict__ dst,
                             int K, int total) {
  int o = blockIdx.x * blockDim.x + threadIdx.x;
  if (o >= total) return;
  int per = K * 64;
  int t = o / per, r = o % per;
  int j    = r & 15;
  int lane = (r >> 4) & 31;
  int fi   = r >> 9;
  int kc = fi >> 2, nt = fi & 3;
  int n = nt * 16 + (lane & 15);
  int k = kc * 32 + (j >> 3) * 16 + (lane >> 4) * 8 + (j & 7);
  dst[o] = (bf16)src[(size_t)t * per + k * 64 + n];
}

// ---------------------------------------------------------------------------
// Prep: dequantize h, mirror state into d_out, make bf16 copy of h.
// ---------------------------------------------------------------------------
__global__ void k_prep(const float* __restrict__ pos_in, const float* __restrict__ vel_in,
                       const float* __restrict__ h_in, const float* __restrict__ g_in,
                       const float* __restrict__ dq_s, const float* __restrict__ dq_t,
                       float* __restrict__ h_o, float* __restrict__ pos_o,
                       float* __restrict__ vel_o, float* __restrict__ g_o,
                       bf16* __restrict__ hbf) {
  int idx = blockIdx.x * blockDim.x + threadIdx.x;
  if (idx >= NH) return;
  int c = idx & 63, node = idx >> 6;
  float hv = h_in[idx] * __expf(dq_s[c]) + dq_t[c];
  h_o[idx] = hv;
  hbf[idx] = (bf16)hv;
  g_o[idx] = g_in[idx];
  if (c < 3) {
    pos_o[node * 3 + c] = pos_in[node * 3 + c];
    vel_o[node * 3 + c] = vel_in[node * 3 + c];
  }
}

__global__ void k_ldj_init(const float* __restrict__ dq_s, float* __restrict__ ldj) {
  float s = 0.f;
  for (int i = 0; i < H; ++i) s += dq_s[i];
  ldj[0] = (float)N_NODES * s;
}

__global__ void k_zero(float* __restrict__ p, int n) {
  int i = blockIdx.x * blockDim.x + threadIdx.x;
  if (i < n) p[i] = 0.f;
}

// ---------------------------------------------------------------------------
// Node MLP: Q = silu(silu(h@W1+b1)@W2+b2)@W3+b3 ; ldj += sum(Q)
// ---------------------------------------------------------------------------
__global__ void __launch_bounds__(WAVES * 32)
k_node(const bf16* __restrict__ hbf,
       const bf16* __restrict__ W1f, const bf16* __restrict__ W2f,
       const float* __restrict__ b1, const float* __restrict__ b2,
       const float* __restrict__ W3, const float* __restrict__ b3,
       float* __restrict__ Q, float* __restrict__ ldj, int nTiles) {
  __shared__ __align__(32) bf16 ldsY[WAVES][16 * 64];
  __shared__ __align__(32) bf16 ldsZ[WAVES][16 * 64];
  int wave = threadIdx.x >> 5, lane = threadIdx.x & 31;
  int tile = blockIdx.x * WAVES + wave;
  if (tile >= nTiles) return;
  int m = lane & 15, hi = lane >> 4, ncol = lane & 15;
  int node = tile * 16 + m;
  const bf16* row = hbf + (size_t)node * 64;

  v16bf a0 = load_a_frag(row, 0, hi);
  v16bf a1 = load_a_frag(row, 32, hi);

#pragma unroll
  for (int nt = 0; nt < 4; ++nt) {            // layer 1
    v8f acc = {};
    acc = wmma_bf16(a0, load_b_frag(W1f, 0, nt, lane), acc);
    acc = wmma_bf16(a1, load_b_frag(W1f, 1, nt, lane), acc);
    float bb = b1[nt * 16 + ncol];
#pragma unroll
    for (int r = 0; r < 8; ++r)
      ldsY[wave][(hi * 8 + r) * 64 + nt * 16 + ncol] = (bf16)silu_f(acc[r] + bb);
  }
  wait_lds();

  const bf16* yrow = &ldsY[wave][m * 64];
  v16bf y0 = load_a_frag(yrow, 0, hi);
  v16bf y1 = load_a_frag(yrow, 32, hi);
#pragma unroll
  for (int nt = 0; nt < 4; ++nt) {            // layer 2
    v8f acc = {};
    acc = wmma_bf16(y0, load_b_frag(W2f, 0, nt, lane), acc);
    acc = wmma_bf16(y1, load_b_frag(W2f, 1, nt, lane), acc);
    float bb = b2[nt * 16 + ncol];
#pragma unroll
    for (int r = 0; r < 8; ++r)
      ldsZ[wave][(hi * 8 + r) * 64 + nt * 16 + ncol] = (bf16)silu_f(acc[r] + bb);
  }
  wait_lds();

  if (lane < 16) {                            // layer 3: 64 -> 1
    float q = b3[0];
#pragma unroll 8
    for (int k = 0; k < 64; ++k) q += (float)ldsZ[wave][m * 64 + k] * W3[k];
    Q[node] = q;
    atomicAdd(ldj, q);
  }
}

// ---------------------------------------------------------------------------
// Edge kernel: msg = silu(hp@eW1+eb1)@eW2+eb2 scatter-> G[dst];
//              a = silu(hp@qW1+qb1)@qW2+qb2; F -/+= 2*a*exp(-r2)*d (analytic grad)
// ---------------------------------------------------------------------------
__global__ void __launch_bounds__(WAVES * 32)
k_edge(const bf16* __restrict__ hbf,
       const int* __restrict__ srcI, const int* __restrict__ dstI,
       const float* __restrict__ pos,
       const bf16* __restrict__ eW1f, const float* __restrict__ eb1,
       const bf16* __restrict__ eW2f, const float* __restrict__ eb2,
       const bf16* __restrict__ qW1f, const float* __restrict__ qb1,
       const float* __restrict__ qW2, const float* __restrict__ qb2,
       float* __restrict__ G, float* __restrict__ F, int nTiles) {
  __shared__ __align__(32) bf16 ldsY[WAVES][16 * 64];
  __shared__ __align__(32) bf16 ldsZ[WAVES][16 * 64];
  int wave = threadIdx.x >> 5, lane = threadIdx.x & 31;
  int tile = blockIdx.x * WAVES + wave;
  if (tile >= nTiles) return;
  int m = lane & 15, hi = lane >> 4, ncol = lane & 15;
  int e = tile * 16 + m;
  int s = srcI[e], d = dstI[e];
  __builtin_prefetch(srcI + e + 16, 0, 1);    // global_prefetch_b8 next tile
  __builtin_prefetch(dstI + e + 16, 0, 1);
  const bf16* rowS = hbf + (size_t)s * 64;
  const bf16* rowD = hbf + (size_t)d * 64;

  // hp = [h[src] | h[dst]] : K = 128 -> 4 A fragments
  v16bf hp0 = load_a_frag(rowS, 0, hi);
  v16bf hp1 = load_a_frag(rowS, 32, hi);
  v16bf hp2 = load_a_frag(rowD, 0, hi);
  v16bf hp3 = load_a_frag(rowD, 32, hi);

#pragma unroll
  for (int nt = 0; nt < 4; ++nt) {            // edge MLP layer 1 (K=128)
    v8f acc = {};
    acc = wmma_bf16(hp0, load_b_frag(eW1f, 0, nt, lane), acc);
    acc = wmma_bf16(hp1, load_b_frag(eW1f, 1, nt, lane), acc);
    acc = wmma_bf16(hp2, load_b_frag(eW1f, 2, nt, lane), acc);
    acc = wmma_bf16(hp3, load_b_frag(eW1f, 3, nt, lane), acc);
    float bb = eb1[nt * 16 + ncol];
#pragma unroll
    for (int r = 0; r < 8; ++r)
      ldsY[wave][(hi * 8 + r) * 64 + nt * 16 + ncol] = (bf16)silu_f(acc[r] + bb);
  }
  wait_lds();

  const bf16* yrow = &ldsY[wave][m * 64];
  v16bf y0 = load_a_frag(yrow, 0, hi);
  v16bf y1 = load_a_frag(yrow, 32, hi);
  int drow[8];
#pragma unroll
  for (int r = 0; r < 8; ++r) drow[r] = dstI[tile * 16 + hi * 8 + r];
#pragma unroll
  for (int nt = 0; nt < 4; ++nt) {            // edge MLP layer 2 + scatter-sum
    v8f acc = {};
    acc = wmma_bf16(y0, load_b_frag(eW2f, 0, nt, lane), acc);
    acc = wmma_bf16(y1, load_b_frag(eW2f, 1, nt, lane), acc);
    float bb = eb2[nt * 16 + ncol];
#pragma unroll
    for (int r = 0; r < 8; ++r)
      atomicAdd(&G[(size_t)drow[r] * 64 + nt * 16 + ncol], acc[r] + bb);
  }

#pragma unroll
  for (int nt = 0; nt < 4; ++nt) {            // energy pair MLP layer 1 (reuses hp)
    v8f acc = {};
    acc = wmma_bf16(hp0, load_b_frag(qW1f, 0, nt, lane), acc);
    acc = wmma_bf16(hp1, load_b_frag(qW1f, 1, nt, lane), acc);
    acc = wmma_bf16(hp2, load_b_frag(qW1f, 2, nt, lane), acc);
    acc = wmma_bf16(hp3, load_b_frag(qW1f, 3, nt, lane), acc);
    float bb = qb1[nt * 16 + ncol];
#pragma unroll
    for (int r = 0; r < 8; ++r)
      ldsZ[wave][(hi * 8 + r) * 64 + nt * 16 + ncol] = (bf16)silu_f(acc[r] + bb);
  }
  wait_lds();

  if (lane < 16) {                            // a = y2 . qW2 + qb2 ; forces
    float a = qb2[0];
#pragma unroll 8
    for (int k = 0; k < 64; ++k) a += (float)ldsZ[wave][m * 64 + k] * qW2[k];
    float dd[3], r2 = 0.f;
#pragma unroll
    for (int c = 0; c < 3; ++c) {
      float x = pos[(size_t)d * 3 + c] - pos[(size_t)s * 3 + c];
      x -= BOX * rintf(x * (1.0f / BOX));     // minimum-image PBC
      dd[c] = x;
      r2 += x * x;
    }
    float w = 2.0f * a * __expf(-r2);
#pragma unroll
    for (int c = 0; c < 3; ++c) {
      float f = w * dd[c];
      atomicAdd(&F[(size_t)d * 3 + c],  f);
      atomicAdd(&F[(size_t)s * 3 + c], -f);
    }
  }
}

// ---------------------------------------------------------------------------
// Integrator update: vel = exp(Q)*vel + F*DT; pos = mod(pos+vel*DT, BOX);
// g += G*DT; h += g*DT; refresh bf16 h.
// ---------------------------------------------------------------------------
__global__ void k_update(float* __restrict__ h, float* __restrict__ pos,
                         float* __restrict__ vel, float* __restrict__ g,
                         const float* __restrict__ G, const float* __restrict__ F,
                         const float* __restrict__ Q, bf16* __restrict__ hbf) {
  int idx = blockIdx.x * blockDim.x + threadIdx.x;
  if (idx >= NH) return;
  int c = idx & 63, node = idx >> 6;
  float gv = g[idx] + G[idx] * DT;
  g[idx] = gv;
  float hv = h[idx] + gv * DT;
  h[idx] = hv;
  hbf[idx] = (bf16)hv;
  if (c < 3) {
    size_t i3 = (size_t)node * 3 + c;
    float v = __expf(Q[node]) * vel[i3] + F[i3] * DT;
    vel[i3] = v;
    float p = pos[i3] + v * DT;
    p -= floorf(p * (1.0f / BOX)) * BOX;      // wrap into [0, BOX)
    pos[i3] = p;
  }
}

// ---------------------------------------------------------------------------
extern "C" void kernel_launch(void* const* d_in, const int* in_sizes, int n_in,
                              void* d_out, int out_size, void* d_ws, size_t ws_size,
                              hipStream_t stream) {
  (void)in_sizes; (void)n_in; (void)out_size; (void)ws_size;
  const float* pos_in = (const float*)d_in[0];
  const float* vel_in = (const float*)d_in[1];
  const float* h_in   = (const float*)d_in[2];
  const float* g_in   = (const float*)d_in[3];
  const int*   edges  = (const int*)d_in[4];
  const float* dq_s   = (const float*)d_in[5];
  const float* dq_t   = (const float*)d_in[6];
  const float* nW1 = (const float*)d_in[7];  const float* nb1 = (const float*)d_in[8];
  const float* nW2 = (const float*)d_in[9];  const float* nb2 = (const float*)d_in[10];
  const float* nW3 = (const float*)d_in[11]; const float* nb3 = (const float*)d_in[12];
  const float* eW1 = (const float*)d_in[13]; const float* eb1 = (const float*)d_in[14];
  const float* eW2 = (const float*)d_in[15]; const float* eb2 = (const float*)d_in[16];
  const float* qW1 = (const float*)d_in[17]; const float* qb1 = (const float*)d_in[18];
  const float* qW2 = (const float*)d_in[19]; const float* qb2 = (const float*)d_in[20];

  // d_out = [h | pos | vel | g | ldj]
  float* h_o   = (float*)d_out;
  float* pos_o = h_o + NH;
  float* vel_o = pos_o + N3;
  float* g_o   = vel_o + N3;
  float* ldj_o = g_o + NH;

  // workspace carve (G and F contiguous so one zero pass clears both)
  float* Gw  = (float*)d_ws;                    // NH
  float* Fw  = Gw + (size_t)NH;                 // N3
  float* Qw  = Fw + (size_t)N3;                 // N
  bf16*  hbf = (bf16*)(Qw + N_NODES);           // NH bf16
  bf16* eW1f = hbf + (size_t)NH;                // T*128*64
  bf16* eW2f = eW1f + (size_t)T_ITER * 128 * 64;
  bf16* qW1f = eW2f + (size_t)T_ITER * 64 * 64;
  bf16* nW1f = qW1f + (size_t)T_ITER * 128 * 64;
  bf16* nW2f = nW1f + (size_t)T_ITER * 64 * 64;

  const int srcOff = 0, dstOff = N_EDGES;

  // one-time prep
  k_prep<<<(NH + 255) / 256, 256, 0, stream>>>(pos_in, vel_in, h_in, g_in, dq_s, dq_t,
                                               h_o, pos_o, vel_o, g_o, hbf);
  k_ldj_init<<<1, 1, 0, stream>>>(dq_s, ldj_o);

  auto wconv = [&](const float* src, bf16* dst, int K) {
    int total = T_ITER * K * 64;
    k_wconv_frag<<<(total + 255) / 256, 256, 0, stream>>>(src, dst, K, total);
  };
  wconv(eW1, eW1f, 128);
  wconv(eW2, eW2f, 64);
  wconv(qW1, qW1f, 128);
  wconv(nW1, nW1f, 64);
  wconv(nW2, nW2f, 64);

  const int nodeBlocks = (NODE_TILES + WAVES - 1) / WAVES;
  const int edgeBlocks = (EDGE_TILES + WAVES - 1) / WAVES;

  for (int t = 0; t < T_ITER; ++t) {
    k_zero<<<(NH + N3 + 255) / 256, 256, 0, stream>>>(Gw, NH + N3);
    k_node<<<nodeBlocks, WAVES * 32, 0, stream>>>(
        hbf, nW1f + (size_t)t * 64 * 64, nW2f + (size_t)t * 64 * 64,
        nb1 + t * H, nb2 + t * H, nW3 + t * H, nb3 + t, Qw, ldj_o, NODE_TILES);
    k_edge<<<edgeBlocks, WAVES * 32, 0, stream>>>(
        hbf, edges + srcOff, edges + dstOff, pos_o,
        eW1f + (size_t)t * 128 * 64, eb1 + t * H,
        eW2f + (size_t)t * 64 * 64,  eb2 + t * H,
        qW1f + (size_t)t * 128 * 64, qb1 + t * H,
        qW2 + t * H, qb2 + t, Gw, Fw, EDGE_TILES);
    k_update<<<(NH + 255) / 256, 256, 0, stream>>>(h_o, pos_o, vel_o, g_o, Gw, Fw, Qw, hbf);
  }
}
